// ViTBlock_22574348108544
// MI455X (gfx1250) — matrix operand order repaired and checked by reading, not compiled, not run on hardware
//
#include <hip/hip_runtime.h>
#include <hip/hip_bf16.h>
#include <math.h>

#define B_ 32
#define N_ 256
#define D_ 768
#define H_ 12
#define M_ 3072

typedef __attribute__((ext_vector_type(16))) __bf16 v16bf;
typedef __attribute__((ext_vector_type(8)))  float  v8f;

union FragAB { v16bf v; uint4 q[2]; unsigned short h[16]; };
union FragC  { v8f v; float f[8]; };

__device__ __forceinline__ unsigned short f2bf(float x) {
  union { float f; unsigned u; } c; c.f = x;
  return (unsigned short)((c.u + 0x7FFFu + ((c.u >> 16) & 1u)) >> 16);
}

// ---------------- f32 -> bf16 convert ----------------
__global__ void cvt_bf16_kernel(const float* __restrict__ in,
                                unsigned short* __restrict__ out, long n) {
  long i = (long)blockIdx.x * blockDim.x + threadIdx.x;
  long stride = (long)gridDim.x * blockDim.x;
  for (; i < n; i += stride) out[i] = f2bf(in[i]);
}

// ---------------- generic bf16 WMMA GEMM:  C[z] = A[z] * Bt[z]^T ----------------
enum { GF_BIAS_COL = 1, GF_BIAS_ROW = 2, GF_GELU = 4, GF_OUT_BF16 = 8, GF_STORE_T = 16 };

template <int FLAGS>
__global__ __launch_bounds__(256)
void gemm_bf16_kernel(const unsigned short* __restrict__ A, long aBatch, int aDiv,
                      const unsigned short* __restrict__ Bt, long bBatch, int bMod,
                      const float* __restrict__ bias, long biasBatch, int biasMod,
                      void* __restrict__ Cout, long cBatch,
                      int Mdim, int Ndim, int Kdim, int lda, int ldb, int ldc) {
  (void)Mdim; (void)Ndim;
  __shared__ __align__(16) unsigned short sA[128 * 40];
  __shared__ __align__(16) unsigned short sB[128 * 40];

  int z = blockIdx.z;
  const unsigned short* Ab = A + (long)(z / aDiv) * aBatch;
  const unsigned short* Bb = Bt + (long)(bMod > 0 ? (z % bMod) : z) * bBatch;
  const float* biasP = bias + (long)(biasMod > 0 ? (z % biasMod) : 0) * biasBatch;

  int tileM = blockIdx.y * 128;
  int tileN = blockIdx.x * 128;

  int tid = threadIdx.x;
  int lane = tid & 31;
  int waveId = tid >> 5;
  int lm = lane & 15, lh = lane >> 4;
  int wRow = (waveId >> 2) * 64;  // 0 or 64
  int wCol = (waveId & 3) * 32;   // 0,32,64,96

  FragC acc[4][2];
#pragma unroll
  for (int r = 0; r < 4; ++r)
#pragma unroll
    for (int c = 0; c < 2; ++c)
#pragma unroll
      for (int e = 0; e < 8; ++e) acc[r][c].f[e] = 0.f;

  int ldrow = tid >> 2;
  int ldcol = (tid & 3) * 8;

  for (int k0 = 0; k0 < Kdim; k0 += 32) {
    __syncthreads();
#pragma unroll
    for (int i = 0; i < 2; ++i) {
      int row = ldrow + i * 64;
      *(uint4*)&sA[row * 40 + ldcol] =
          *(const uint4*)&Ab[(long)(tileM + row) * lda + k0 + ldcol];
      *(uint4*)&sB[row * 40 + ldcol] =
          *(const uint4*)&Bb[(long)(tileN + row) * ldb + k0 + ldcol];
    }
    __syncthreads();

    FragAB a[4], b[2];
#pragma unroll
    for (int r = 0; r < 4; ++r) {
      const unsigned short* base = &sA[(wRow + r * 16 + lm) * 40];
      a[r].q[0] = *(const uint4*)(base + 8 * lh);
      a[r].q[1] = *(const uint4*)(base + 16 + 8 * lh);
    }
#pragma unroll
    for (int c = 0; c < 2; ++c) {
      const unsigned short* base = &sB[(wCol + c * 16 + lm) * 40];
      b[c].q[0] = *(const uint4*)(base + 8 * lh);
      b[c].q[1] = *(const uint4*)(base + 16 + 8 * lh);
    }
#pragma unroll
    for (int r = 0; r < 4; ++r)
#pragma unroll
      for (int c = 0; c < 2; ++c)
        acc[r][c].v = __builtin_amdgcn_wmma_f32_16x16x32_bf16(
            false, a[r].v, false, b[c].v, (short)0, acc[r][c].v, false, false);
  }

#pragma unroll
  for (int r = 0; r < 4; ++r)
#pragma unroll
    for (int c = 0; c < 2; ++c)
#pragma unroll
      for (int e = 0; e < 8; ++e) {
        int row = tileM + wRow + r * 16 + 8 * lh + e;
        int col = tileN + wCol + c * 16 + lm;
        float v = acc[r][c].f[e];
        if (FLAGS & GF_BIAS_COL) v += biasP[col];
        if (FLAGS & GF_BIAS_ROW) v += biasP[row];
        if (FLAGS & GF_GELU) v = 0.5f * v * (1.f + erff(v * 0.70710678118f));
        long idx = (FLAGS & GF_STORE_T) ? ((long)col * ldc + row)
                                        : ((long)row * ldc + col);
        if (FLAGS & GF_OUT_BF16)
          ((unsigned short*)Cout)[(long)z * cBatch + idx] = f2bf(v);
        else
          ((float*)Cout)[(long)z * cBatch + idx] = v;
      }
}

// ---------------- attention: S=QK^T -> softmax (in 266KB LDS) -> P*V ----------------
__global__ __launch_bounds__(256)
void attn_kernel(const unsigned short* __restrict__ Q,
                 const unsigned short* __restrict__ Kmat,
                 const unsigned short* __restrict__ Vt,
                 unsigned short* __restrict__ attT) {
  extern __shared__ char smem[];
  float* sS = (float*)smem;                        // [256][260] f32 scores
  float* rinv = (float*)(smem + 256 * 260 * 4);    // [256] 1/rowsum

  int z = blockIdx.x;
  int b = z / H_, h = z % H_;
  const unsigned short* q = Q + (long)z * N_ * D_;
  const unsigned short* k = Kmat + (long)z * N_ * D_;
  const unsigned short* vT = Vt + (long)z * D_ * N_;  // [768][256] = V^T

  int tid = threadIdx.x;
  int lane = tid & 31, waveId = tid >> 5;
  int lm = lane & 15, lh = lane >> 4;
  int wRow = waveId * 32;  // 8 waves x 32 rows

  // Phase 1: raw scores into LDS
  for (int cb = 0; cb < N_; cb += 64) {
    FragC acc[2][4];
#pragma unroll
    for (int r = 0; r < 2; ++r)
#pragma unroll
      for (int c = 0; c < 4; ++c)
#pragma unroll
        for (int e = 0; e < 8; ++e) acc[r][c].f[e] = 0.f;

    for (int kk = 0; kk < D_; kk += 32) {
      FragAB a[2], bf[4];
#pragma unroll
      for (int r = 0; r < 2; ++r) {
        const unsigned short* base = q + (long)(wRow + r * 16 + lm) * D_ + kk;
        a[r].q[0] = *(const uint4*)(base + 8 * lh);
        a[r].q[1] = *(const uint4*)(base + 16 + 8 * lh);
      }
#pragma unroll
      for (int c = 0; c < 4; ++c) {
        const unsigned short* base = k + (long)(cb + c * 16 + lm) * D_ + kk;
        bf[c].q[0] = *(const uint4*)(base + 8 * lh);
        bf[c].q[1] = *(const uint4*)(base + 16 + 8 * lh);
      }
#pragma unroll
      for (int r = 0; r < 2; ++r)
#pragma unroll
        for (int c = 0; c < 4; ++c)
          acc[r][c].v = __builtin_amdgcn_wmma_f32_16x16x32_bf16(
              false, a[r].v, false, bf[c].v, (short)0, acc[r][c].v, false, false);
    }
#pragma unroll
    for (int r = 0; r < 2; ++r)
#pragma unroll
      for (int c = 0; c < 4; ++c)
#pragma unroll
        for (int e = 0; e < 8; ++e) {
          int row = wRow + r * 16 + 8 * lh + e;
          int col = cb + c * 16 + lm;
          sS[row * 260 + col] = acc[r][c].f[e];
        }
  }
  __syncthreads();

  // Phase 2: per-row softmax, bf16 P written in place (byte 2c < byte 4c, same row base)
  {
    const float scale = 0.06262242910851495f;  // 1/sqrt(255)
    float* Sr = sS + tid * 260;
    unsigned short* Pr = (unsigned short*)Sr;
    float mx = -1e30f;
    for (int c = 0; c < N_; ++c) mx = fmaxf(mx, Sr[c]);
    float sum = 0.f;
    for (int c = 0; c < N_; ++c) {
      float e = __expf((Sr[c] - mx) * scale);
      sum += e;
      Pr[c] = f2bf(e);
    }
    rinv[tid] = 1.f / sum;
  }
  __syncthreads();

  // Phase 3: att = P*V, scaled by 1/rowsum, stored transposed: attT[b][d][h*N + tok]
  const unsigned short* Pbase = (const unsigned short*)smem;  // row stride 520
  long outBase = ((long)b * D_) * (H_ * N_) + (long)h * N_;
  for (int cb = 0; cb < D_; cb += 64) {
    FragC acc[2][4];
#pragma unroll
    for (int r = 0; r < 2; ++r)
#pragma unroll
      for (int c = 0; c < 4; ++c)
#pragma unroll
        for (int e = 0; e < 8; ++e) acc[r][c].f[e] = 0.f;

    for (int kk = 0; kk < N_; kk += 32) {
      FragAB a[2], bf[4];
#pragma unroll
      for (int r = 0; r < 2; ++r) {
        const unsigned short* base = Pbase + (long)(wRow + r * 16 + lm) * 520 + kk;
        a[r].q[0] = *(const uint4*)(base + 8 * lh);
        a[r].q[1] = *(const uint4*)(base + 16 + 8 * lh);
      }
#pragma unroll
      for (int c = 0; c < 4; ++c) {
        const unsigned short* base = vT + (long)(cb + c * 16 + lm) * N_ + kk;
        bf[c].q[0] = *(const uint4*)(base + 8 * lh);
        bf[c].q[1] = *(const uint4*)(base + 16 + 8 * lh);
      }
#pragma unroll
      for (int r = 0; r < 2; ++r)
#pragma unroll
        for (int c = 0; c < 4; ++c)
          acc[r][c].v = __builtin_amdgcn_wmma_f32_16x16x32_bf16(
              false, a[r].v, false, bf[c].v, (short)0, acc[r][c].v, false, false);
    }
#pragma unroll
    for (int r = 0; r < 2; ++r)
#pragma unroll
      for (int c = 0; c < 4; ++c)
#pragma unroll
        for (int e = 0; e < 8; ++e) {
          int row = wRow + r * 16 + 8 * lh + e;
          int col = cb + c * 16 + lm;
          float v = acc[r][c].f[e] * rinv[row];
          attT[outBase + (long)col * (H_ * N_) + row] = f2bf(v);
        }
  }
}

// ---------------- fused residual add + LayerNorm (one wave per row) ----------------
__global__ __launch_bounds__(256)
void add_ln_kernel(const float* __restrict__ X, const float* __restrict__ Y,
                   const float* __restrict__ g, const float* __restrict__ be,
                   float* __restrict__ outF, unsigned short* __restrict__ outBF,
                   int cols) {
  int wave = threadIdx.x >> 5, lane = threadIdx.x & 31;
  long row = (long)blockIdx.x * 8 + wave;
  const float* xr = X + row * cols;
  const float* yr = Y + row * cols;
  float s = 0.f, ss = 0.f;
  for (int c = lane; c < cols; c += 32) {
    float v = xr[c] + yr[c];
    s += v;
    ss += v * v;
  }
#pragma unroll
  for (int off = 16; off; off >>= 1) {
    s += __shfl_xor(s, off, 32);
    ss += __shfl_xor(ss, off, 32);
  }
  float mean = s / cols;
  float var = ss / cols - mean * mean;
  float rs = rsqrtf(var + 1e-5f);
  for (int c = lane; c < cols; c += 32) {
    float v = (xr[c] + yr[c] - mean) * rs * g[c] + be[c];
    outF[row * cols + c] = v;
    if (outBF) outBF[row * cols + c] = f2bf(v);
  }
}

// ---------------- host orchestration ----------------
extern "C" void kernel_launch(void* const* d_in, const int* in_sizes, int n_in,
                              void* d_out, int out_size, void* d_ws, size_t ws_size,
                              hipStream_t stream) {
  (void)in_sizes; (void)n_in; (void)out_size; (void)ws_size;
  const float* x     = (const float*)d_in[0];
  const float* Wq    = (const float*)d_in[1];
  const float* bq    = (const float*)d_in[2];
  const float* Wk    = (const float*)d_in[3];
  const float* bk    = (const float*)d_in[4];
  const float* Wv    = (const float*)d_in[5];
  const float* bv    = (const float*)d_in[6];
  const float* Wmap  = (const float*)d_in[7];
  const float* bmap  = (const float*)d_in[8];
  const float* g1    = (const float*)d_in[9];
  const float* beta1 = (const float*)d_in[10];
  const float* W1    = (const float*)d_in[11];
  const float* b1    = (const float*)d_in[12];
  const float* W2    = (const float*)d_in[13];
  const float* b2    = (const float*)d_in[14];
  const float* g2    = (const float*)d_in[15];
  const float* beta2 = (const float*)d_in[16];

  char* ws = (char*)d_ws;
  size_t off = 0;
  auto alloc = [&](size_t bytes) {
    off = (off + 255) & ~(size_t)255;
    size_t o = off;
    off += bytes;
    return o;
  };

  unsigned short* xb    = (unsigned short*)(ws + alloc((size_t)B_ * N_ * D_ * 2));
  unsigned short* Wqb   = (unsigned short*)(ws + alloc((size_t)H_ * D_ * D_ * 2));
  unsigned short* Wkb   = (unsigned short*)(ws + alloc((size_t)H_ * D_ * D_ * 2));
  unsigned short* Wvb   = (unsigned short*)(ws + alloc((size_t)H_ * D_ * D_ * 2));
  unsigned short* Wmapb = (unsigned short*)(ws + alloc((size_t)N_ * H_ * N_ * 2));
  unsigned short* W1b   = (unsigned short*)(ws + alloc((size_t)M_ * D_ * 2));
  unsigned short* W2b   = (unsigned short*)(ws + alloc((size_t)D_ * M_ * 2));
  unsigned short* qb    = (unsigned short*)(ws + alloc((size_t)B_ * H_ * N_ * D_ * 2));
  unsigned short* kb    = (unsigned short*)(ws + alloc((size_t)B_ * H_ * N_ * D_ * 2));
  unsigned short* vTb   = (unsigned short*)(ws + alloc((size_t)B_ * H_ * D_ * N_ * 2));
  unsigned short* attTb = (unsigned short*)(ws + alloc((size_t)B_ * D_ * H_ * N_ * 2));
  float*          msa   = (float*)(ws + alloc((size_t)B_ * N_ * D_ * 4));
  float*          outF  = (float*)(ws + alloc((size_t)B_ * N_ * D_ * 4));
  unsigned short* outBF = (unsigned short*)(ws + alloc((size_t)B_ * N_ * D_ * 2));
  unsigned short* hid   = (unsigned short*)(ws + alloc((size_t)B_ * N_ * M_ * 2));
  float*          mlp   = (float*)(ws + alloc((size_t)B_ * N_ * D_ * 4));

  auto cvt = [&](const float* src, unsigned short* dst, long n) {
    int blocks = (int)((n + 1023) / 1024);
    cvt_bf16_kernel<<<blocks, 256, 0, stream>>>(src, dst, n);
  };
  cvt(x, xb, (long)B_ * N_ * D_);
  cvt(Wq, Wqb, (long)H_ * D_ * D_);
  cvt(Wk, Wkb, (long)H_ * D_ * D_);
  cvt(Wv, Wvb, (long)H_ * D_ * D_);
  cvt(Wmap, Wmapb, (long)N_ * H_ * N_);
  cvt(W1, W1b, (long)M_ * D_);
  cvt(W2, W2b, (long)D_ * M_);

  // QKV projections: z = b*H + h ; q/k normal bf16, v stored transposed [768][256]
  dim3 gQKV(D_ / 128, N_ / 128, B_ * H_);
  gemm_bf16_kernel<GF_BIAS_COL | GF_OUT_BF16><<<gQKV, 256, 0, stream>>>(
      xb, (long)N_ * D_, H_, Wqb, (long)D_ * D_, H_, bq, (long)D_, H_,
      qb, (long)N_ * D_, N_, D_, D_, D_, D_, D_);
  gemm_bf16_kernel<GF_BIAS_COL | GF_OUT_BF16><<<gQKV, 256, 0, stream>>>(
      xb, (long)N_ * D_, H_, Wkb, (long)D_ * D_, H_, bk, (long)D_, H_,
      kb, (long)N_ * D_, N_, D_, D_, D_, D_, D_);
  gemm_bf16_kernel<GF_BIAS_COL | GF_OUT_BF16 | GF_STORE_T><<<gQKV, 256, 0, stream>>>(
      xb, (long)N_ * D_, H_, Wvb, (long)D_ * D_, H_, bv, (long)D_, H_,
      vTb, (long)N_ * D_, N_, D_, D_, D_, D_, N_);

  // attention: one workgroup per (b,h), 266KB dynamic LDS for full score matrix
  size_t attSmem = (size_t)256 * 260 * 4 + 256 * 4;
  attn_kernel<<<B_ * H_, 256, attSmem, stream>>>(qb, kb, vTb, attTb);

  // head-merge linear over token axis: msa[b] = Wmap * attT[b]^T  (bias per row)
  dim3 gMap(D_ / 128, N_ / 128, B_);
  gemm_bf16_kernel<GF_BIAS_ROW><<<gMap, 256, 0, stream>>>(
      Wmapb, 0L, 1, attTb, (long)D_ * H_ * N_, 0, bmap, 0L, 0,
      msa, (long)N_ * D_, N_, D_, H_ * N_, H_ * N_, H_ * N_, D_);

  // LN1(x + msa) -> outF (f32) + outBF (bf16)
  add_ln_kernel<<<(B_ * N_) / 8, 256, 0, stream>>>(x, msa, g1, beta1, outF, outBF, D_);

  // MLP1: hid = gelu(out @ W1^T + b1)  [256 x 3072] bf16
  dim3 gM1(M_ / 128, N_ / 128, B_);
  gemm_bf16_kernel<GF_BIAS_COL | GF_GELU | GF_OUT_BF16><<<gM1, 256, 0, stream>>>(
      outBF, (long)N_ * D_, 1, W1b, 0L, 1, b1, 0L, 0,
      hid, (long)N_ * M_, N_, M_, D_, D_, D_, M_);

  // MLP2: mlp = hid @ W2^T + b2  [256 x 768] f32
  dim3 gM2(D_ / 128, N_ / 128, B_);
  gemm_bf16_kernel<GF_BIAS_COL><<<gM2, 256, 0, stream>>>(
      hid, (long)N_ * M_, 1, W2b, 0L, 1, b2, 0L, 0,
      mlp, (long)N_ * D_, N_, D_, M_, M_, M_, D_);

  // LN2(out + mlp) -> d_out (f32)
  add_ln_kernel<<<(B_ * N_) / 8, 256, 0, stream>>>(outF, mlp, g2, beta2,
                                                   (float*)d_out, nullptr, D_);
}
